// SingleLayer_47107201303276
// MI455X (gfx1250) — matrix operand order repaired and checked by reading, not compiled, not run on hardware
//
#include <hip/hip_runtime.h>
#include <hip/hip_bf16.h>

typedef __attribute__((ext_vector_type(16))) _Float16 v16h;
typedef __attribute__((ext_vector_type(8)))  _Float16 v8h;
typedef __attribute__((ext_vector_type(8)))  float    v8f;

#define BATCH   8192
#define CIN     448
#define LLEN    15
#define CL      (CIN*LLEN)      // 6720
#define OUTC    480
#define OCL     (OUTC*LLEN)     // 7200
#define NGROW   32
#define INCH    64
#define NGRP    12
#define KDIM    (CIN*3)         // 1344
#define NCHUNK  (KDIM/32)       // 42
#define NSEG    16
#define SEGB    (BATCH/NSEG)    // 512
#define NCTILE  (CL/64)         // 105
#define BN_EPSF 1e-5f
#define ROWH    456             // LDS row stride in halves (16B aligned, conflict-free)
#define HROWS   17              // rows: l=-1..15  (row0 zero, row16 zero)
#define WPB     4               // waves per block in k4

// ---- workspace layout (bytes) ----
constexpr size_t OFF_PS  = 0;                                   // NSEG*CL f32 partial sums
constexpr size_t OFF_PQ  = OFF_PS + (size_t)NSEG*CL*4;          // NSEG*CL f32 partial sumsq
constexpr size_t OFF_CS  = OFF_PQ + (size_t)NSEG*CL*4;          // CL f32 column sums
constexpr size_t OFF_CQ  = OFF_CS + (size_t)CL*4;               // CL f32 column sumsq
constexpr size_t OFF_AC  = OFF_CQ + (size_t)CL*4;               // CIN f32 affine A (source-indexed)
constexpr size_t OFF_BC  = OFF_AC + (size_t)CIN*4;              // CIN f32 affine B (source-indexed)
constexpr size_t OFF_DST = OFF_BC + (size_t)CIN*4;              // CIN i32 dest column (source-indexed)
constexpr size_t OFF_W   = ((OFF_DST + (size_t)CIN*4 + 31)/32)*32; // 42*2*32*16 f16 swizzled weights

// ---------------------------------------------------------------------------
// Pass 1: per-(channel,pos) batch sums + sumsq, segmented over batch.
// ---------------------------------------------------------------------------
__global__ void __launch_bounds__(256) k1_partials(const float* __restrict__ x,
                                                   void* __restrict__ wsv) {
    float* pS = (float*)((char*)wsv + OFF_PS);
    float* pQ = (float*)((char*)wsv + OFF_PQ);
    int tile = blockIdx.x % NCTILE;
    int seg  = blockIdx.x / NCTILE;
    int co   = threadIdx.x & 63;
    int br   = threadIdx.x >> 6;
    int col  = tile*64 + co;
    float acc = 0.f, accq = 0.f;
    int b0 = seg*SEGB;
    for (int b = b0 + br; b < b0 + SEGB; b += 4) {
        float v = x[(size_t)b*CL + col];
        acc += v; accq += v*v;
    }
    __shared__ float red[256];
    red[threadIdx.x] = acc;
    __syncthreads();
    if (threadIdx.x < 64) {
        float s = red[threadIdx.x] + red[threadIdx.x+64] + red[threadIdx.x+128] + red[threadIdx.x+192];
        pS[seg*CL + col] = s;
    }
    __syncthreads();
    red[threadIdx.x] = accq;
    __syncthreads();
    if (threadIdx.x < 64) {
        float s = red[threadIdx.x] + red[threadIdx.x+64] + red[threadIdx.x+128] + red[threadIdx.x+192];
        pQ[seg*CL + col] = s;
    }
}

// ---------------------------------------------------------------------------
// Pass 2 (single block): reduce partials, gating scalars, fold BN into a
// per-channel affine (stored SOURCE-indexed, permutation is an involution),
// and pre-swizzle im2col conv weights into per-lane WMMA B-fragment order.
// ---------------------------------------------------------------------------
__global__ void __launch_bounds__(1024) k2_prep(const float* __restrict__ bn_gamma,
                                                const float* __restrict__ bn_beta,
                                                const float* __restrict__ conv_w,
                                                const float* __restrict__ L1w,
                                                const float* __restrict__ L1b,
                                                const float* __restrict__ L2w,
                                                void* __restrict__ wsv) {
    char* w = (char*)wsv;
    float* pS   = (float*)(w + OFF_PS);
    float* pQ   = (float*)(w + OFF_PQ);
    float* cS   = (float*)(w + OFF_CS);
    float* cQ   = (float*)(w + OFF_CQ);
    float* Ac   = (float*)(w + OFF_AC);
    float* Bc   = (float*)(w + OFF_BC);
    int*   dstC = (int*)  (w + OFF_DST);
    _Float16* wim2 = (_Float16*)(w + OFF_W);

    __shared__ float mu_s[CIN], var_s[CIN];
    __shared__ float sg_s[NGRP];
    __shared__ float wg_s[NGRP-1];
    __shared__ float dot_s, w0_s;
    int tid = threadIdx.x, nt = blockDim.x;

    // A: reduce batch-segment partials to full column sums
    for (int col = tid; col < CL; col += nt) {
        float s = 0.f, q = 0.f;
        for (int g = 0; g < NSEG; ++g) { s += pS[g*CL+col]; q += pQ[g*CL+col]; }
        cS[col] = s; cQ[col] = q;
    }
    __syncthreads();
    // B: per-channel mean / variance of x
    for (int c = tid; c < CIN; c += nt) {
        float s = 0.f, q = 0.f;
        for (int l = 0; l < LLEN; ++l) { s += cS[c*LLEN+l]; q += cQ[c*LLEN+l]; }
        const float inv = 1.0f/(float)(BATCH*LLEN);
        float mu = s*inv;
        mu_s[c]  = mu;
        var_s[c] = q*inv - mu*mu;
    }
    __syncthreads();
    // C: gating dot products (deterministic)
    const float invB = 1.0f/(float)BATCH;
    if (tid < NGRP) {
        float s = 0.f;
        for (int jj = 0; jj < NGROW*LLEN; ++jj) {
            int ch = INCH + tid*NGROW + jj/LLEN;
            s += (cS[ch*LLEN + jj%LLEN]*invB) * L2w[jj];
        }
        sg_s[tid] = s;
    } else if (tid == NGRP) {
        float d = 0.f;
        for (int j = 0; j < INCH*LLEN; ++j) d += (cS[j]*invB)*L1w[j];
        dot_s = d;
    }
    __syncthreads();
    if (tid == 0) {
        float slast = sg_s[NGRP-1];
        w0_s = 1.f/(1.f + __expf(-(slast + dot_s + L1b[0])));
        for (int g = 0; g < NGRP-1; ++g)
            wg_s[g] = 1.f/(1.f + __expf(-(sg_s[g] + slast)));
    }
    __syncthreads();
    // D: fold scale+BN+gamma/beta into affine; store indexed by SOURCE channel
    for (int cx = tid; cx < CIN; cx += nt) {
        int src; float sc;
        if (cx < INCH) { src = cx; sc = w0_s; }
        else if (cx < INCH + (NGRP-1)*NGROW) {
            int g = (cx - INCH) >> 5;
            int r = (cx - INCH) & 31;
            int sg = (NGRP-2) - g;           // reversed group order
            src = INCH + sg*NGROW + r;
            sc  = wg_s[sg];
        } else { src = cx; sc = 1.f; }
        float rinv = rsqrtf(sc*sc*var_s[src] + BN_EPSF);
        Ac[src]   = sc * bn_gamma[cx] * rinv;              // h[cx] = Ac*x[src] + Bc
        Bc[src]   = bn_beta[cx] - sc*mu_s[src]*bn_gamma[cx]*rinv;
        dstC[src] = cx;                                    // column of h for this x channel
    }
    // E: pre-swizzled im2col weights, f16, per-lane B-fragment order:
    //    halves i=0..15 of lane ln hold K = chunk*32 + (ln>=16?8:0) + (i&7) + (i>=8?16:0), N = t*16 + (ln&15)
    for (int idx = tid; idx < NCHUNK*2*32*16; idx += nt) {
        int i     = idx & 15;
        int lane  = (idx >> 4) & 31;
        int t     = (idx >> 9) & 1;
        int chunk = idx >> 10;
        int K  = chunk*32 + ((lane & 16) ? 8 : 0) + (i & 7) + ((i & 8) ? 16 : 0);
        int N  = t*16 + (lane & 15);
        int dl = (K >= 2*CIN) ? 2 : ((K >= CIN) ? 1 : 0);
        int cx = K - dl*CIN;
        wim2[idx] = (_Float16)conv_w[N*KDIM + cx*3 + dl];
    }
}

// ---------------------------------------------------------------------------
// Pass 3a: passthrough copy x -> out[:, :448] (float4, peak-BW)
// ---------------------------------------------------------------------------
__global__ void __launch_bounds__(256) k3_copy(const float4* __restrict__ x4,
                                               float4* __restrict__ out4) {
    unsigned i = blockIdx.x*blockDim.x + threadIdx.x;
    const unsigned n4 = (unsigned)BATCH*(CL/4);
    if (i >= n4) return;
    unsigned b = i / (CL/4);
    unsigned r = i - b*(CL/4);
    out4[(size_t)b*(OCL/4) + r] = x4[i];
}

// ---------------------------------------------------------------------------
// Pass 3b: conv as WMMA GEMM, one wave per sample.
// Phase 1: h (448x15, f16) computed once into LDS as [l+1][cx], zero guard rows.
// Phase 2: A fragments = two 16B ds_load_b128 per chunk (8-K runs never cross
//          a dl segment since 448 % 8 == 0); B fragments pre-swizzled in ws.
// ---------------------------------------------------------------------------
__global__ void __launch_bounds__(WPB*32) k4_conv_wmma(const float* __restrict__ x,
                                                       const void* __restrict__ wsv,
                                                       float* __restrict__ out) {
    const char* w = (const char*)wsv;
    const float* A2  = (const float*)(w + OFF_AC);
    const float* B2  = (const float*)(w + OFF_BC);
    const int*   dst = (const int*)  (w + OFF_DST);
    const _Float16* wim2 = (const _Float16*)(w + OFF_W);

    __shared__ _Float16 hh[WPB * HROWS * ROWH];   // 62,016 bytes

    int wslot = threadIdx.x >> 5;
    int lane  = threadIdx.x & 31;
    int wave  = blockIdx.x * WPB + wslot;         // batch sample
    if (wave >= BATCH) return;                    // wave-uniform
    _Float16* hhw = hh + wslot * (HROWS * ROWH);

    // zero guard rows: row 0 (l=-1, also used by the M==15 pad lane), row 16 (l=15)
    for (int i = lane; i < ROWH; i += 32) {
        hhw[i] = (_Float16)0.f;
        hhw[16*ROWH + i] = (_Float16)0.f;
    }

    // Phase 1: coalesced read of the sample, affine+relu+f16, scatter into LDS
    const float* xb = x + (size_t)wave * CL;
    for (int idx = lane; idx < CL; idx += 32) {
        int c = idx / LLEN;
        int l = idx - c*LLEN;
        float v = fmaxf(A2[c]*xb[idx] + B2[c], 0.f);
        hhw[(l+1)*ROWH + dst[c]] = (_Float16)v;
    }
    __syncthreads();

    int  M  = lane & 15;
    bool hi = lane >= 16;
    bool valid = (M < 15);

    union { v16h v; v8h h[2]; } a;
    v8f c0 = {}; v8f c1 = {};
    for (int chunk = 0; chunk < NCHUNK; ++chunk) {
        int k0  = chunk*32 + (hi ? 8 : 0);
        int k1  = k0 + 16;
        int dl0 = (k0 >= 2*CIN) ? 2 : ((k0 >= CIN) ? 1 : 0);
        int dl1 = (k1 >= 2*CIN) ? 2 : ((k1 >= CIN) ? 1 : 0);
        int cx0 = k0 - dl0*CIN;
        int cx1 = k1 - dl1*CIN;
        int r0  = valid ? (M + dl0) : 0;          // row = l+1 = M+dl ; pad lane -> zero row
        int r1  = valid ? (M + dl1) : 0;
        a.h[0] = *(const v8h*)(hhw + r0*ROWH + cx0);
        a.h[1] = *(const v8h*)(hhw + r1*ROWH + cx1);
        const v16h b0 = *(const v16h*)(wim2 + ((size_t)(chunk*2+0)*32 + lane)*16);
        const v16h b1 = *(const v16h*)(wim2 + ((size_t)(chunk*2+1)*32 + lane)*16);
        c0 = __builtin_amdgcn_wmma_f32_16x16x32_f16(false, a.v, false, b0, (short)0, c0, false, false);
        c1 = __builtin_amdgcn_wmma_f32_16x16x32_f16(false, a.v, false, b1, (short)0, c1, false, false);
    }

    float* ob = out + (size_t)wave*OCL + (size_t)CIN*LLEN;
    int N = lane & 15;
    #pragma unroll
    for (int r = 0; r < 8; ++r) {
        int Mr = (hi ? 8 : 0) + r;                // D layout: lanes<16 -> M=0..7, else 8..15
        if (Mr < LLEN) {
            ob[N*LLEN + Mr]      = c0[r];
            ob[(N+16)*LLEN + Mr] = c1[r];
        }
    }
}

// ---------------------------------------------------------------------------
extern "C" void kernel_launch(void* const* d_in, const int* in_sizes, int n_in,
                              void* d_out, int out_size, void* d_ws, size_t ws_size,
                              hipStream_t stream) {
    const float* x      = (const float*)d_in[0];
    const float* gamma  = (const float*)d_in[1];
    const float* beta   = (const float*)d_in[2];
    const float* conv_w = (const float*)d_in[3];
    const float* L1w    = (const float*)d_in[4];
    const float* L1b    = (const float*)d_in[5];
    const float* L2w    = (const float*)d_in[6];
    float* out = (float*)d_out;

    k1_partials<<<NCTILE*NSEG, 256, 0, stream>>>(x, d_ws);
    k2_prep<<<1, 1024, 0, stream>>>(gamma, beta, conv_w, L1w, L1b, L2w, d_ws);
    unsigned n4 = (unsigned)BATCH*(CL/4);
    k3_copy<<<(n4 + 255)/256, 256, 0, stream>>>((const float4*)x, (float4*)d_out);
    k4_conv_wmma<<<BATCH/WPB, WPB*32, 0, stream>>>(x, d_ws, out);
}
